// MAC_30554397344312
// MI455X (gfx1250) — matrix operand order repaired and checked by reading, not compiled, not run on hardware
//
#include <hip/hip_runtime.h>
#include <hip/hip_bf16.h>

typedef __attribute__((ext_vector_type(16))) _Float16 v16h;
typedef __attribute__((ext_vector_type(8)))  _Float16 v8h;
typedef __attribute__((ext_vector_type(4)))  _Float16 v4h;
typedef __attribute__((ext_vector_type(8)))  float    v8f;
typedef __attribute__((ext_vector_type(4)))  float    v4f;

#define DD   256      // hidden dim
#define NAG  64       // agents per batch
#define LDA  264      // padded LDS row stride in halfs (528B = 4-bank shift/row)

// ---------------------------------------------------------------------------
// Kernel 1: convert the three 256x256 fp32 weight matrices to f16 once.
// Packed layout in workspace: [W_act | W_sum | W_head], 65536 halfs each.
// ---------------------------------------------------------------------------
__global__ void MAC_cvt_weights(const float* __restrict__ wa,
                                const float* __restrict__ ws,
                                const float* __restrict__ wh,
                                _Float16* __restrict__ o) {
    int i = blockIdx.x * blockDim.x + threadIdx.x;      // 0..65535
    o[i]               = (_Float16)wa[i];
    o[i + DD * DD]     = (_Float16)ws[i];
    o[i + 2 * DD * DD] = (_Float16)wh[i];
}

// ---------------------------------------------------------------------------
// One 64x256 x 256x256 GEMM (C = A_lds @ W^T) into 8 WMMA tiles per wave.
// A fragment: K contiguous per lane (k = 16*(lane/16) + s), matching B -> two ds_load_b128 per lane.
// B fragment: row-major W[c,k] serves as B in this code's k order (N = lane%16 = out-channel,
//             K contiguous per lane) -> one 32B contiguous global load.
// B fragments are double-buffered in registers one k-step ahead so no WMMA
// waits on a just-issued global load.
// ---------------------------------------------------------------------------
__device__ __forceinline__ void gemm_acc(const _Float16 (*A)[LDA],
                                         const _Float16* __restrict__ W,
                                         int m0, int nb, int lane,
                                         v8f acc[8]) {
    const int row = m0 + (lane & 15);
    const int kAo = (lane >> 4) ? 16 : 0;   // A: half-wave K offset
    const int kBo = (lane >> 4) ? 16 : 0;   // B: half-wave K offset
    const _Float16* __restrict__ Wl = W + (nb + (lane & 15)) * DD + kBo;

    v16h b[8];
#pragma unroll
    for (int t = 0; t < 8; ++t)
        b[t] = *(const v16h*)(Wl + t * 16 * DD);        // prefetch k-step 0

#pragma unroll
    for (int k0 = 0; k0 < DD; k0 += 32) {
        // A fragment for this k-step (LDS, short latency)
        union { v16h v; v8h h[2]; } ua;
        ua.h[0] = *(const v8h*)&A[row][k0 + kAo];
        ua.h[1] = *(const v8h*)&A[row][k0 + kAo + 8];
        const v16h a = ua.v;

        // issue next k-step's B loads before consuming current ones
        v16h bn[8];
        if (k0 + 32 < DD) {
#pragma unroll
            for (int t = 0; t < 8; ++t)
                bn[t] = *(const v16h*)(Wl + t * 16 * DD + (k0 + 32));
        }

#pragma unroll
        for (int t = 0; t < 8; ++t)
            acc[t] = __builtin_amdgcn_wmma_f32_16x16x32_f16(
                false, a, false, b[t], (short)0, acc[t], false, false);

        if (k0 + 32 < DD) {
#pragma unroll
            for (int t = 0; t < 8; ++t) b[t] = bn[t];   // SSA copy, folds away
        }
    }
}

// ---------------------------------------------------------------------------
// Kernel 2: one workgroup per batch; fused 3-layer pipeline with the
// exclusive-mean agent aggregation between layer 1 and layer 2.
// ---------------------------------------------------------------------------
__global__ __launch_bounds__(256)
void MAC_fused(const float* __restrict__ hidden,
               const float* __restrict__ b_act,
               const float* __restrict__ b_sum,
               const float* __restrict__ b_head,
               const _Float16* __restrict__ wpack,
               float* __restrict__ out) {
    __shared__ _Float16 Abuf[NAG][LDA];   // ~33 KB activation staging (f16)

    const int tid  = threadIdx.x;
    const int lane = tid & 31;
    const int wave = tid >> 5;                 // 8 waves
    const int m0   = (wave >> 1) * 16;         // 4 row blocks
    const int nb   = (wave & 1) * 128;         // 2 col halves
    const int batch = blockIdx.x;

    // D-tile addressing for epilogues (C/D layout: M = v + 8*(lane>=16), N = lane%16)
    const int colL = nb + (lane & 15);
    const int rowL = m0 + ((lane >> 4) << 3);

    // ---- stage hidden tile (64x256 fp32 -> f16 in LDS), coalesced v4 loads
    const float* src = hidden + (size_t)batch * NAG * DD;
    for (int e = tid * 4; e < NAG * DD; e += 256 * 4) {
        const int r = e >> 8, c = e & (DD - 1);
        const v4f f = *(const v4f*)(src + e);
        v4h h;
        h.x = (_Float16)f.x; h.y = (_Float16)f.y;
        h.z = (_Float16)f.z; h.w = (_Float16)f.w;
        *(v4h*)&Abuf[r][c] = h;
    }
    __syncthreads();

    const v8f zacc = {};
    v8f acc[8];

    // ================= layer 1: comm = hidden @ W_act^T + b_act ============
#pragma unroll
    for (int t = 0; t < 8; ++t) acc[t] = zacc;
    gemm_acc(Abuf, wpack, m0, nb, lane, acc);
    __syncthreads();                           // all reads of hidden done
#pragma unroll
    for (int t = 0; t < 8; ++t) {
        const float bv = b_act[colL + 16 * t];
#pragma unroll
        for (int v = 0; v < 8; ++v)
            Abuf[rowL + v][colL + 16 * t] = (_Float16)(acc[t][v] + bv);
    }
    __syncthreads();

    // ===== exclusive-mean aggregation: agg[m,d] = (S_d - comm[m,d]) / 63 ====
    {
        float s = 0.f;
        for (int m = 0; m < NAG; ++m) s += (float)Abuf[m][tid];
        const float inv = 1.0f / (float)(NAG - 1);
        for (int m = 0; m < NAG; ++m)
            Abuf[m][tid] = (_Float16)((s - (float)Abuf[m][tid]) * inv);
    }
    __syncthreads();

    // ================= layer 2: x = agg @ W_sum^T + b_sum ==================
#pragma unroll
    for (int t = 0; t < 8; ++t) acc[t] = zacc;
    gemm_acc(Abuf, wpack + DD * DD, m0, nb, lane, acc);
    __syncthreads();                           // all reads of agg done
#pragma unroll
    for (int t = 0; t < 8; ++t) {
        const float bv = b_sum[colL + 16 * t];
#pragma unroll
        for (int v = 0; v < 8; ++v)
            Abuf[rowL + v][colL + 16 * t] = (_Float16)(acc[t][v] + bv);
    }
    __syncthreads();

    // ============ layer 3: out = relu(x @ W_head^T + b_head) ===============
#pragma unroll
    for (int t = 0; t < 8; ++t) acc[t] = zacc;
    gemm_acc(Abuf, wpack + 2 * DD * DD, m0, nb, lane, acc);
#pragma unroll
    for (int t = 0; t < 8; ++t) {
        const float bv = b_head[colL + 16 * t];
#pragma unroll
        for (int v = 0; v < 8; ++v) {
            float x = acc[t][v] + bv;
            x = x > 0.f ? x : 0.f;
            out[((size_t)batch * NAG + rowL + v) * DD + colL + 16 * t] = x;
        }
    }
}

// ---------------------------------------------------------------------------
extern "C" void kernel_launch(void* const* d_in, const int* in_sizes, int n_in,
                              void* d_out, int out_size, void* d_ws, size_t ws_size,
                              hipStream_t stream) {
    const float* hidden = (const float*)d_in[0];
    const float* W_act  = (const float*)d_in[1];
    const float* b_act  = (const float*)d_in[2];
    const float* W_sum  = (const float*)d_in[3];
    const float* b_sum  = (const float*)d_in[4];
    const float* W_head = (const float*)d_in[5];
    const float* b_head = (const float*)d_in[6];
    float* out = (float*)d_out;

    _Float16* wpack = (_Float16*)d_ws;                 // needs 3*65536*2 = 384 KB

    const int nbatch = in_sizes[0] / (NAG * DD);       // = 256

    MAC_cvt_weights<<<(DD * DD) / 256, 256, 0, stream>>>(W_act, W_sum, W_head, wpack);
    MAC_fused<<<nbatch, 256, 0, stream>>>(hidden, b_act, b_sum, b_head, wpack, out);
}
